// Attention2_43739946942860
// MI455X (gfx1250) — matrix operand order repaired and checked by reading, not compiled, not run on hardware
//
#include <hip/hip_runtime.h>
#include <hip/hip_bf16.h>

// ---------------------------------------------------------------------------
// Problem constants (from reference): B=8, LQ=2048, LK=2048, H=1024
// ---------------------------------------------------------------------------
#define B_  8
#define LQ_ 2048
#define LK_ 2048
#define H_  1024

typedef _Float16 v8h  __attribute__((ext_vector_type(8)));
typedef _Float16 v16h __attribute__((ext_vector_type(16)));
typedef float    v8f  __attribute__((ext_vector_type(8)));
typedef int      v4i  __attribute__((ext_vector_type(4)));

#if __has_builtin(__builtin_amdgcn_global_load_async_to_lds_b128)
#define HAS_ASYNC_LDS 1
#else
#define HAS_ASYNC_LDS 0
#endif

static __device__ __forceinline__ v16h cat16(v8h lo, v8h hi) {
  v16h r;
#pragma unroll
  for (int i = 0; i < 8; ++i) { r[i] = lo[i]; r[i + 8] = hi[i]; }
  return r;
}

// 16-byte global -> LDS copy: async on CDNA5 when available (ASYNCcnt-tracked,
// no VGPR staging), else synchronous through registers.
static __device__ __forceinline__ void copy16_g2l(const _Float16* g, _Float16* l) {
#if HAS_ASYNC_LDS
  auto gv = (__attribute__((address_space(1))) void*)g;   // generic -> global AS
  auto lv = (__attribute__((address_space(3))) void*)l;   // generic -> LDS AS
  __builtin_amdgcn_global_load_async_to_lds_b128(
      (__attribute__((address_space(1))) v4i*)gv,
      (__attribute__((address_space(3))) v4i*)lv, 0, 0);
#else
  *(v8h*)l = *(const v8h*)g;
#endif
}

template <int N>
static __device__ __forceinline__ void wait_async() {
#if HAS_ASYNC_LDS
#if __has_builtin(__builtin_amdgcn_s_wait_asynccnt)
  __builtin_amdgcn_s_wait_asynccnt(N);
#else
  if constexpr (N == 0) asm volatile("s_wait_asynccnt 0x0" ::: "memory");
  else                  asm volatile("s_wait_asynccnt 0x4" ::: "memory");
#endif
#endif
}

// ---------------------------------------------------------------------------
// Elementwise f32 -> f16 cast
// ---------------------------------------------------------------------------
__global__ __launch_bounds__(256) void cast_f32_f16(const float* __restrict__ in,
                                                    _Float16* __restrict__ out,
                                                    long n) {
  long i = (long)blockIdx.x * blockDim.x + threadIdx.x;
  if (i < n) out[i] = (_Float16)in[i];
}

// ---------------------------------------------------------------------------
// Tiled transpose + cast: in [R, C] f32 row-major -> out [C, R] f16 (batched)
// R, C multiples of 32.
// ---------------------------------------------------------------------------
__global__ __launch_bounds__(256) void transpose_cast(const float* __restrict__ in,
                                                      _Float16* __restrict__ out,
                                                      int R, int C,
                                                      long inStride, long outStride) {
  __shared__ float tile[32][33];
  const int b = blockIdx.z;
  const float* I = in + (long)b * inStride;
  _Float16* O = out + (long)b * outStride;

  const int x  = blockIdx.x * 32 + threadIdx.x;   // input col
  const int y0 = blockIdx.y * 32;                 // input row base
#pragma unroll
  for (int j = 0; j < 4; ++j)
    tile[threadIdx.y + j * 8][threadIdx.x] = I[(long)(y0 + threadIdx.y + j * 8) * C + x];
  __syncthreads();
  const int ox  = y0 + threadIdx.x;               // output col (= input row)
  const int oy0 = blockIdx.x * 32;                // output row base (= input col)
#pragma unroll
  for (int j = 0; j < 4; ++j)
    O[(long)(oy0 + threadIdx.y + j * 8) * R + ox] =
        (_Float16)tile[threadIdx.x][threadIdx.y + j * 8];
}

// ---------------------------------------------------------------------------
// NT GEMM via v_wmma_f32_16x16x32_f16:
//   D[m][n] = sum_k A[m][k] * Bm[n][k]      (both row-major, contiguous K)
// Block tile 128x128, 256 threads = 8 waves, K-step 32, double-buffered LDS
// staged with async global->LDS copies.
// MODE 0: f16 output.  MODE 1: f32 output with mask (==0 -> -inf).  MODE 2: f32.
// M,N multiples of 128; K multiple of 32.
// ---------------------------------------------------------------------------
#define BM 128
#define BN 128
#define BK 32
#define LDT 40   // LDS row stride in halves (padded vs 32 to break bank conflicts)

template <int MODE>
__global__ __launch_bounds__(256) void gemm_nt(
    const _Float16* __restrict__ Aall, long aStride,
    const _Float16* __restrict__ Ball, long bStride,
    float* __restrict__ D32all, long dStride,
    _Float16* __restrict__ D16all,
    const int* __restrict__ maskAll,
    int M, int N, int K) {
  (void)M;
  const int b = blockIdx.z;
  const _Float16* A  = Aall + (long)b * aStride;
  const _Float16* Bm = Ball + (long)b * bStride;

  const int tid  = threadIdx.x;
  const int lane = tid & 31;
  const int wave = tid >> 5;
  const int wm   = wave & 3;   // 4 row groups of 32
  const int wn   = wave >> 2;  // 2 col groups of 64
  const int mBase = blockIdx.y * BM;
  const int nBase = blockIdx.x * BN;

  __shared__ alignas(16) _Float16 sA[2][BM * LDT];
  __shared__ alignas(16) _Float16 sB[2][BN * LDT];

  v8f acc[2][4];
#pragma unroll
  for (int i = 0; i < 2; ++i)
#pragma unroll
    for (int j = 0; j < 4; ++j) acc[i][j] = (v8f){0.f, 0.f, 0.f, 0.f, 0.f, 0.f, 0.f, 0.f};

  // staging: 2 threads per row, each covers 16 halves (two b128 chunks)
  const int ldRow = tid >> 1;
  const int ldSeg = (tid & 1) * 16;

  // fragment lane mapping per CDNA5 16-bit WMMA VGPR layouts (wave32)
  const int fm  = lane & 15;
  const int ko  = (lane < 16) ? 0 : 8;   // A: K = {ko..ko+7} U {16+ko..16+ko+7}
  const int fn  = lane & 15;
  const int klo = (lane < 16) ? 0 : 8;   // B: K = {klo..klo+7} U {16+klo..16+klo+7}

  const _Float16* gA = A  + (long)(mBase + ldRow) * K + ldSeg;
  const _Float16* gB = Bm + (long)(nBase + ldRow) * K + ldSeg;

  // prologue: stage tile 0 into buffer 0 (4 async b128 copies per thread)
  {
    _Float16* la = &sA[0][ldRow * LDT + ldSeg];
    _Float16* lb = &sB[0][ldRow * LDT + ldSeg];
    copy16_g2l(gA,     la);
    copy16_g2l(gA + 8, la + 8);
    copy16_g2l(gB,     lb);
    copy16_g2l(gB + 8, lb + 8);
  }

  int buf = 0;
  for (int kk = 0; kk < K; kk += BK, buf ^= 1) {
    const bool more = (kk + BK) < K;
    if (more) {  // stage next tile into the other buffer while computing this one
      _Float16* la = &sA[buf ^ 1][ldRow * LDT + ldSeg];
      _Float16* lb = &sB[buf ^ 1][ldRow * LDT + ldSeg];
      const _Float16* na = gA + kk + BK;
      const _Float16* nb = gB + kk + BK;
      copy16_g2l(na,     la);
      copy16_g2l(na + 8, la + 8);
      copy16_g2l(nb,     lb);
      copy16_g2l(nb + 8, lb + 8);
      wait_async<4>();   // async loads retire in order: current tile has landed
    } else {
      wait_async<0>();
    }
    __syncthreads();     // current tile visible to all waves

    v16h afrag[2];
#pragma unroll
    for (int i = 0; i < 2; ++i) {
      const int m = wm * 32 + i * 16 + fm;
      v8h lo = *(const v8h*)(&sA[buf][m * LDT + ko]);
      v8h hi = *(const v8h*)(&sA[buf][m * LDT + 16 + ko]);
      afrag[i] = cat16(lo, hi);
    }
    v16h bfrag[4];
#pragma unroll
    for (int j = 0; j < 4; ++j) {
      const int n = wn * 64 + j * 16 + fn;
      v8h lo = *(const v8h*)(&sB[buf][n * LDT + klo]);
      v8h hi = *(const v8h*)(&sB[buf][n * LDT + klo + 16]);
      bfrag[j] = cat16(lo, hi);
    }
#pragma unroll
    for (int i = 0; i < 2; ++i)
#pragma unroll
      for (int j = 0; j < 4; ++j)
        acc[i][j] = __builtin_amdgcn_wmma_f32_16x16x32_f16(
            false, afrag[i], false, bfrag[j], (short)0, acc[i][j], false, false);

    __syncthreads();     // reads done before next iteration overwrites this buffer
  }

  // epilogue: C/D layout -> row = base + r + (lane<16?0:8), col = base + (lane&15)
  const int rOff = (lane < 16) ? 0 : 8;
#pragma unroll
  for (int i = 0; i < 2; ++i) {
#pragma unroll
    for (int j = 0; j < 4; ++j) {
      const int col = nBase + wn * 64 + j * 16 + (lane & 15);
      int mval = 1;
      if (MODE == 1) mval = maskAll[(long)b * N + col];
#pragma unroll
      for (int r = 0; r < 8; ++r) {
        const int row = mBase + wm * 32 + i * 16 + rOff + r;
        const float v = acc[i][j][r];
        if (MODE == 0) {
          D16all[(long)b * dStride + (long)row * N + col] = (_Float16)v;
        } else if (MODE == 1) {
          D32all[(long)b * dStride + (long)row * N + col] =
              mval ? v : -__builtin_inff();
        } else {
          D32all[(long)b * dStride + (long)row * N + col] = v;
        }
      }
    }
  }
}

// ---------------------------------------------------------------------------
// Row softmax + sigmoid over masked scores (in place in probs_softmax region).
// One 256-thread block per (b,q) row of length LK=2048 (8 elems/thread).
// Emits: probs_softmax (f32, in place), probs_sigmoid (f32),
//        Ph = f16(probs_softmax), Phat = f16(score > 0 ? 1 : 0).
// ---------------------------------------------------------------------------
__global__ __launch_bounds__(256) void softmax_row(
    float* __restrict__ scores, float* __restrict__ sigm,
    _Float16* __restrict__ Ph, _Float16* __restrict__ Phat) {
  const long row = blockIdx.x;
  float* s = scores + row * (long)LK_;
  const int t = threadIdx.x;

  float v[8];
#pragma unroll
  for (int i = 0; i < 8; ++i) v[i] = s[t + i * 256];

  __shared__ float red[256];
  float m = -__builtin_inff();
#pragma unroll
  for (int i = 0; i < 8; ++i) m = fmaxf(m, v[i]);
  red[t] = m;
  __syncthreads();
  for (int off = 128; off > 0; off >>= 1) {
    if (t < off) red[t] = fmaxf(red[t], red[t + off]);
    __syncthreads();
  }
  m = red[0];
  __syncthreads();

  float e[8];
  float sum = 0.f;
#pragma unroll
  for (int i = 0; i < 8; ++i) { e[i] = expf(v[i] - m); sum += e[i]; }
  red[t] = sum;
  __syncthreads();
  for (int off = 128; off > 0; off >>= 1) {
    if (t < off) red[t] += red[t + off];
    __syncthreads();
  }
  sum = red[0];
  const float inv = 1.0f / sum;

#pragma unroll
  for (int i = 0; i < 8; ++i) {
    const int k = t + i * 256;
    const float p = e[i] * inv;
    s[k] = p;                                       // probs_softmax (in place)
    sigm[row * (long)LK_ + k] = 1.0f / (1.0f + expf(-v[i]));  // sigmoid(-inf)=0
    Ph[row * (long)LK_ + k]   = (_Float16)p;
    Phat[row * (long)LK_ + k] = (v[i] > 0.0f) ? (_Float16)1.0f : (_Float16)0.0f;
  }
}

// ---------------------------------------------------------------------------
// Host-side launch
// ---------------------------------------------------------------------------
extern "C" void kernel_launch(void* const* d_in, const int* in_sizes, int n_in,
                              void* d_out, int out_size, void* d_ws, size_t ws_size,
                              hipStream_t stream) {
  (void)in_sizes; (void)n_in; (void)out_size; (void)ws_size;

  const float* Q = (const float*)d_in[0];   // [B, LQ, H]
  const float* K = (const float*)d_in[1];   // [B, LK, H]
  const float* V = (const float*)d_in[2];   // [B, LK, H]
  const float* W = (const float*)d_in[3];   // [H, H]
  const int*   mask = (const int*)d_in[4];  // [B, LK]

  float* out = (float*)d_out;
  float* outSoft  = out;                                        // [B,LQ,H]
  float* outSig   = out + (long)B_ * LQ_ * H_;                  // [B,LQ,H]
  float* probsS   = out + 2L * B_ * LQ_ * H_;                   // [B,LQ,LK]
  float* probsSig = probsS + (long)B_ * LQ_ * LK_;              // [B,LQ,LK]

  // workspace layout (bytes)
  char* ws = (char*)d_ws;
  const long SZ_QKV16 = (long)B_ * LQ_ * H_ * 2;   // 33.5 MB each
  _Float16* Qh   = (_Float16*)(ws);
  _Float16* Kh   = (_Float16*)(ws + SZ_QKV16);
  _Float16* Vt   = (_Float16*)(ws + 2 * SZ_QKV16);          // [B, H, LK]
  _Float16* Wt   = (_Float16*)(ws + 3 * SZ_QKV16);          // [H, H] (W^T)
  _Float16* Ah   = (_Float16*)(ws + 3 * SZ_QKV16 + (long)H_ * H_ * 2);
  _Float16* Ph   = (_Float16*)((char*)Ah + SZ_QKV16);       // [B, LQ, LK]
  _Float16* Phat = (_Float16*)((char*)Ph + (long)B_ * LQ_ * LK_ * 2);

  const long nQ = (long)B_ * LQ_ * H_;

  // 1) casts / transposes to f16
  cast_f32_f16<<<(nQ + 255) / 256, 256, 0, stream>>>(Q, Qh, nQ);
  cast_f32_f16<<<(nQ + 255) / 256, 256, 0, stream>>>(K, Kh, nQ);
  {
    dim3 g(H_ / 32, H_ / 32, 1), t(32, 8);
    transpose_cast<<<g, t, 0, stream>>>(W, Wt, H_, H_, 0, 0);
  }
  {
    dim3 g(H_ / 32, LK_ / 32, B_), t(32, 8);
    transpose_cast<<<g, t, 0, stream>>>(V, Vt, LK_, H_,
                                        (long)LK_ * H_, (long)H_ * LK_);
  }

  // 2) A = Q @ W   (NT against W^T), f16 out
  {
    dim3 g(H_ / BN, LQ_ / BM, B_);
    gemm_nt<0><<<g, 256, 0, stream>>>(Qh, (long)LQ_ * H_, Wt, 0L,
                                      nullptr, (long)LQ_ * H_, Ah, nullptr,
                                      LQ_, H_, H_);
  }
  // 3) S = A @ K^T with mask -> raw masked scores into probs_softmax region
  {
    dim3 g(LK_ / BN, LQ_ / BM, B_);
    gemm_nt<1><<<g, 256, 0, stream>>>(Ah, (long)LQ_ * H_, Kh, (long)LK_ * H_,
                                      probsS, (long)LQ_ * LK_, nullptr, mask,
                                      LQ_, LK_, H_);
  }
  // 4) row softmax / sigmoid; emits f16 P and hard-mask P̂
  softmax_row<<<B_ * LQ_, 256, 0, stream>>>(probsS, probsSig, Ph, Phat);

  // 5) out_softmax = P @ V  and  out_sigmoid = P̂ @ V  (NT against V^T)
  {
    dim3 g(H_ / BN, LQ_ / BM, B_);
    gemm_nt<2><<<g, 256, 0, stream>>>(Ph, (long)LQ_ * LK_, Vt, (long)H_ * LK_,
                                      outSoft, (long)LQ_ * H_, nullptr, nullptr,
                                      LQ_, H_, LK_);
    gemm_nt<2><<<g, 256, 0, stream>>>(Phat, (long)LQ_ * LK_, Vt, (long)H_ * LK_,
                                      outSig, (long)LQ_ * H_, nullptr, nullptr,
                                      LQ_, H_, LK_);
  }
}
